// GRUDT_58548994179399
// MI455X (gfx1250) — matrix operand order repaired and checked by reading, not compiled, not run on hardware
//
#include <hip/hip_runtime.h>
#include <hip/hip_bf16.h>

typedef __bf16 bf16;
typedef __attribute__((ext_vector_type(16))) __bf16 v16bf;
typedef __attribute__((ext_vector_type(8)))  float  v8f;

#define B_DIM 1024
#define T_DIM 100
#define I_DIM 128
#define H_DIM 256
#define G3H   768
#define BT    (B_DIM * T_DIM)

// ---------- WMMA fragment loaders (CDNA5 16x16x32 bf16 layouts) ----------
// A matrix 16x32 (MxK), 16-bit: lanes 0-15 = M rows, K in {0..7,16..23}+8*half
__device__ __forceinline__ v16bf load_frag_a(const bf16* rowptr, int K0, int half) {
  v16bf v;
  const unsigned int* p = (const unsigned int*)rowptr;
#pragma unroll
  for (int i = 0; i < 8; ++i) {
    int k = K0 + ((i < 4) ? (2 * i) : (8 + 2 * i)) + 8 * half;
    union { unsigned int u; __bf16 h[2]; } cv;
    cv.u = p[k >> 1];
    v[2 * i] = cv.h[0];
    v[2 * i + 1] = cv.h[1];
  }
  return v;
}
// B matrix 32x16 (KxN), 16-bit: lane%16 = N column, K in {0..7,16..23}+8*half
// rowptr points at W[n][*] (i.e. B column n laid out contiguously along K)
__device__ __forceinline__ v16bf load_frag_b(const bf16* rowptr, int K0, int half) {
  v16bf v;
  const unsigned int* p = (const unsigned int*)rowptr;
#pragma unroll
  for (int i = 0; i < 8; ++i) {
    int k = K0 + ((i < 4) ? (2 * i) : (8 + 2 * i)) + 8 * half;
    union { unsigned int u; __bf16 h[2]; } cv;
    cv.u = p[k >> 1];
    v[2 * i] = cv.h[0];
    v[2 * i + 1] = cv.h[1];
  }
  return v;
}

__device__ __forceinline__ float sigf(float x) { return 1.f / (1.f + __expf(-x)); }

// ---------- init: fp32 -> bf16 conversions (weights L2-resident afterwards) ----------
__global__ void k_cvt(const float* __restrict__ s, bf16* __restrict__ d, int n) {
  for (int i = blockIdx.x * blockDim.x + threadIdx.x; i < n; i += gridDim.x * blockDim.x)
    d[i] = (bf16)s[i];
}
// d[c*rows + r] = s[r*cols + c]  (so B-operand K runs contiguously per output column)
__global__ void k_cvt_t(const float* __restrict__ s, bf16* __restrict__ d, int rows, int cols) {
  int n = rows * cols;
  for (int i = blockIdx.x * blockDim.x + threadIdx.x; i < n; i += gridDim.x * blockDim.x) {
    int r = i / cols, c = i % cols;
    d[c * rows + r] = (bf16)s[i];
  }
}

// ---------- Kernel A: all non-recurrent math, batched over (b,t) ----------
__global__ __launch_bounds__(256) void k_pregate(
    const float* __restrict__ x, const float* __restrict__ mask,
    const float* __restrict__ x_mean,
    const float* __restrict__ h_gx, const float* __restrict__ h_gh,
    const float* __restrict__ b_ih,
    const bf16* __restrict__ dt_bf, const bf16* __restrict__ wgxT,
    const bf16* __restrict__ wghT, const bf16* __restrict__ wihT,
    float* __restrict__ gamma_h_ws, float* __restrict__ gi_ws,
    float* __restrict__ xhat_out) {
  extern __shared__ char smem_dyn[];
  bf16* lds_xs = (bf16*)smem_dyn;  // 8 waves * 16*128 bf16 = 32 KB

  const int tid = threadIdx.x;
  const int wave = tid >> 5, lane = tid & 31;
  const int half = lane >> 4, ln = lane & 15;
  const int R0 = blockIdx.x * 128 + wave * 16;  // rows are b*T + t

  const bf16* dtrow = dt_bf + (size_t)(R0 + ln) * I_DIM;
  v16bf a_dt[4];
#pragma unroll
  for (int kf = 0; kf < 4; ++kf) a_dt[kf] = load_frag_a(dtrow, kf * 32, half);

  bf16* my_xs = lds_xs + wave * (16 * I_DIM);

  // gamma_x GEMM + xs elementwise (+ write xhat output)
  for (int nt = 0; nt < 8; ++nt) {
    v8f acc = {};
    const bf16* wrow = wgxT + (size_t)(nt * 16 + ln) * I_DIM;
#pragma unroll
    for (int kf = 0; kf < 4; ++kf) {
      v16bf bfr = load_frag_b(wrow, kf * 32, half);
      acc = __builtin_amdgcn_wmma_f32_16x16x32_bf16(false, a_dt[kf], false, bfr,
                                                    (short)0, acc, false, false);
    }
    const int n = nt * 16 + ln;
#pragma unroll
    for (int r = 0; r < 8; ++r) {
      int m = r + 8 * half;
      int grow = R0 + m;
      int t = grow % T_DIM, b = grow / T_DIM;
      float gx = (t == 0) ? 0.f : __expf(-fmaxf(0.f, acc[r] + h_gx[n]));
      size_t gidx = (size_t)grow * I_DIM + n;
      float xv = x[gidx], mv = mask[gidx];
      float xp = (t == 0) ? 0.f : x[gidx - I_DIM];
      float xs = xv * mv + (1.f - mv) * (gx * xp + (1.f - gx) * x_mean[n]);
      xhat_out[((size_t)t * B_DIM + b) * I_DIM + n] = xs;
      my_xs[m * I_DIM + n] = (bf16)xs;
    }
  }
  __syncthreads();

  // gamma_h GEMM
  for (int nt = 0; nt < 16; ++nt) {
    v8f acc = {};
    const bf16* wrow = wghT + (size_t)(nt * 16 + ln) * I_DIM;
#pragma unroll
    for (int kf = 0; kf < 4; ++kf) {
      v16bf bfr = load_frag_b(wrow, kf * 32, half);
      acc = __builtin_amdgcn_wmma_f32_16x16x32_bf16(false, a_dt[kf], false, bfr,
                                                    (short)0, acc, false, false);
    }
    const int n = nt * 16 + ln;
#pragma unroll
    for (int r = 0; r < 8; ++r) {
      int m = r + 8 * half;
      int grow = R0 + m;
      int t = grow % T_DIM, b = grow / T_DIM;
      float g = __expf(-fmaxf(0.f, acc[r] + h_gh[n]));
      gamma_h_ws[((size_t)t * B_DIM + b) * H_DIM + n] = g;
    }
  }

  // gi = xs @ W_ih^T + b_ih (A frags from LDS-staged bf16 xs)
  const bf16* xsrow = my_xs + ln * I_DIM;
  v16bf a_xs[4];
#pragma unroll
  for (int kf = 0; kf < 4; ++kf) a_xs[kf] = load_frag_a(xsrow, kf * 32, half);
  for (int nt = 0; nt < 48; ++nt) {
    v8f acc = {};
    const bf16* wrow = wihT + (size_t)(nt * 16 + ln) * I_DIM;
#pragma unroll
    for (int kf = 0; kf < 4; ++kf) {
      v16bf bfr = load_frag_b(wrow, kf * 32, half);
      acc = __builtin_amdgcn_wmma_f32_16x16x32_bf16(false, a_xs[kf], false, bfr,
                                                    (short)0, acc, false, false);
    }
    const int n = nt * 16 + ln;
#pragma unroll
    for (int r = 0; r < 8; ++r) {
      int m = r + 8 * half;
      int grow = R0 + m;
      int t = grow % T_DIM, b = grow / T_DIM;
      gi_ws[((size_t)t * B_DIM + b) * G3H + n] = acc[r] + b_ih[n];
    }
  }
}

// ---------- Kernel B: batch-parallel persistent recurrence ----------
// block = 16 batch rows, 8 waves. Wave w owns columns j in [32w, 32w+32):
// it computes the r/z/n gate tiles for those columns (nt = jt, 16+jt, 32+jt),
// so gate math runs directly out of the WMMA accumulators (no gh LDS buffer).
// gamma_h[t] tiles are double-buffered in LDS via global_load_async_to_lds_b128.
__global__ __launch_bounds__(256) void k_recur(
    const float* __restrict__ tp, const float* __restrict__ h0,
    const float* __restrict__ w_t, const float* __restrict__ h_t,
    const float* __restrict__ b_hh,
    const float* __restrict__ W1, const float* __restrict__ b1,
    const float* __restrict__ W2, const float* __restrict__ b2,
    const bf16* __restrict__ whh_bf,
    const float* __restrict__ gamma_h_ws, const float* __restrict__ gi_ws,
    float* __restrict__ out) {
  extern __shared__ char smem_dyn[];
  float* h_f = (float*)smem_dyn;              // 16*256 f32   (16 KB)
  float* gam = h_f + 16 * H_DIM;              // 2 * 16*256 f32 (32 KB, double buffer)
  float* hid = gam + 2 * 16 * H_DIM;          // 16*10 f32
  bf16*  h_b = (bf16*)(hid + 160);            // 16*256 bf16  (8 KB)

  const int tid = threadIdx.x;
  const int wave = tid >> 5, lane = tid & 31;
  const int half = lane >> 4, ln = lane & 15;
  const int b0 = blockIdx.x * 16;

  // init h = broadcast(h0)
  for (int i = tid; i < 16 * H_DIM; i += 256) {
    float v = h0[i & (H_DIM - 1)];
    h_f[i] = v;
    h_b[i] = (bf16)v;
  }

  // hoist per-column constants out of the time loop
  float bh_r[2], bh_z[2], bh_n[2], wtj[2], htj[2];
#pragma unroll
  for (int u = 0; u < 2; ++u) {
    int j = wave * 32 + u * 16 + ln;
    bh_r[u] = b_hh[j];
    bh_z[u] = b_hh[H_DIM + j];
    bh_n[u] = b_hh[2 * H_DIM + j];
    wtj[u] = w_t[j];
    htj[u] = h_t[j];
  }

  // async-prefetch gamma_h tile for t=0 into buffer 0 (each thread owns 4 b128 chunks)
  {
    const float* src = gamma_h_ws + (size_t)b0 * H_DIM;
    unsigned lbase = (unsigned)(uintptr_t)gam;
#pragma unroll
    for (int k = 0; k < 4; ++k) {
      int cid = tid + k * 256;
      asm volatile("global_load_async_to_lds_b128 %0, %1, off"
                   :: "v"(lbase + cid * 16),
                      "v"((unsigned long long)(uintptr_t)(src + cid * 4))
                   : "memory");
    }
  }
  __syncthreads();

  for (int t = 0; t < T_DIM; ++t) {
    // wait for this step's gamma tile (each thread reads only chunks it issued)
    asm volatile("s_wait_asynccnt 0" ::: "memory");
    const float* gcur = gam + (t & 1) * (16 * H_DIM);
#pragma unroll
    for (int k = 0; k < 4; ++k) {
      int cid = tid + k * 256;
#pragma unroll
      for (int e = 0; e < 4; ++e) {
        int i = cid * 4 + e;
        float v = gcur[i] * h_f[i];
        h_f[i] = v;
        h_b[i] = (bf16)v;
      }
    }
    // kick off async DMA of next step's gamma tile; overlaps the WMMA phase
    if (t + 1 < T_DIM) {
      float* gnxt = gam + ((t + 1) & 1) * (16 * H_DIM);
      const float* src = gamma_h_ws + ((size_t)(t + 1) * B_DIM + b0) * H_DIM;
      unsigned lbase = (unsigned)(uintptr_t)gnxt;
#pragma unroll
      for (int k = 0; k < 4; ++k) {
        int cid = tid + k * 256;
        asm volatile("global_load_async_to_lds_b128 %0, %1, off"
                     :: "v"(lbase + cid * 16),
                        "v"((unsigned long long)(uintptr_t)(src + cid * 4))
                     : "memory");
      }
    }
    __syncthreads();

    // A fragments of decayed h (bf16, from LDS)
    v16bf a_h[8];
    const bf16* hrow = h_b + ln * H_DIM;
#pragma unroll
    for (int kf = 0; kf < 8; ++kf) a_h[kf] = load_frag_a(hrow, kf * 32, half);
    __syncthreads();  // everyone done reading h_b before gates overwrite it

    // three gate chains for this wave's 32 columns: 48 WMMAs, W_hh streams from L2
    v8f acc_r[2], acc_z[2], acc_n[2];
#pragma unroll
    for (int u = 0; u < 2; ++u) {
      int jt = wave * 2 + u;  // column tile 0..15
      const bf16* wr_r = whh_bf + (size_t)((jt)*16 + ln) * H_DIM;
      const bf16* wr_z = whh_bf + (size_t)((16 + jt) * 16 + ln) * H_DIM;
      const bf16* wr_n = whh_bf + (size_t)((32 + jt) * 16 + ln) * H_DIM;
      v8f ar = {}, az = {}, an = {};
#pragma unroll
      for (int kf = 0; kf < 8; ++kf) {
        v16bf brf = load_frag_b(wr_r, kf * 32, half);
        ar = __builtin_amdgcn_wmma_f32_16x16x32_bf16(false, a_h[kf], false, brf,
                                                     (short)0, ar, false, false);
        v16bf bzf = load_frag_b(wr_z, kf * 32, half);
        az = __builtin_amdgcn_wmma_f32_16x16x32_bf16(false, a_h[kf], false, bzf,
                                                     (short)0, az, false, false);
        v16bf bnf = load_frag_b(wr_n, kf * 32, half);
        an = __builtin_amdgcn_wmma_f32_16x16x32_bf16(false, a_h[kf], false, bnf,
                                                     (short)0, an, false, false);
      }
      acc_r[u] = ar;
      acc_z[u] = az;
      acc_n[u] = an;
    }

    // gate math straight out of the accumulators (lane owns (m, j) triplets)
#pragma unroll
    for (int u = 0; u < 2; ++u) {
      int j = wave * 32 + u * 16 + ln;
#pragma unroll
      for (int rr = 0; rr < 8; ++rr) {
        int m = rr + 8 * half;
        int b = b0 + m;
        size_t grow = ((size_t)t * B_DIM + b) * G3H;
        float i_r = gi_ws[grow + j];
        float i_z = gi_ws[grow + H_DIM + j];
        float i_n = gi_ws[grow + 2 * H_DIM + j];
        float r = sigf(i_r + acc_r[u][rr] + bh_r[u]);
        float z = sigf(i_z + acc_z[u][rr] + bh_z[u]);
        float nn = tanhf(i_n + r * (acc_n[u][rr] + bh_n[u]));
        float hp = h_f[m * H_DIM + j];
        float gt = sigf(tp[(size_t)b * T_DIM + t] * wtj[u] + htj[u]);
        float h1 = gt * ((1.f - z) * nn + z * hp);
        h_f[m * H_DIM + j] = h1;
        h_b[m * H_DIM + j] = (bf16)h1;
      }
    }
    __syncthreads();

    // tiny MLP head: hid = sigmoid(h1 @ W1^T + b1), logits = sigmoid(hid @ W2^T + b2), softmax
    if (tid < 160) {
      int m = tid / 10, o = tid % 10;
      const float* hr = h_f + m * H_DIM;
      const float* wr = W1 + o * H_DIM;
      float s = b1[o];
#pragma unroll 4
      for (int j = 0; j < H_DIM; ++j) s += wr[j] * hr[j];
      hid[tid] = sigf(s);
    }
    __syncthreads();
    if (tid < 16) {
      int m = tid, b = b0 + m;
      float l0 = b2[0], l1 = b2[1];
#pragma unroll
      for (int o = 0; o < 10; ++o) {
        l0 += W2[o] * hid[m * 10 + o];
        l1 += W2[10 + o] * hid[m * 10 + o];
      }
      l0 = sigf(l0);
      l1 = sigf(l1);
      float mx = fmaxf(l0, l1);
      float e0 = __expf(l0 - mx), e1 = __expf(l1 - mx), inv = 1.f / (e0 + e1);
      size_t ob = ((size_t)t * B_DIM + b) * 2;
      out[ob] = e0 * inv;
      out[ob + 1] = e1 * inv;
    }
    __syncthreads();
  }
}

extern "C" void kernel_launch(void* const* d_in, const int* in_sizes, int n_in,
                              void* d_out, int out_size, void* d_ws, size_t ws_size,
                              hipStream_t stream) {
  (void)in_sizes; (void)n_in; (void)out_size; (void)ws_size;
  const float* x      = (const float*)d_in[0];
  const float* dt     = (const float*)d_in[1];
  const float* mask   = (const float*)d_in[2];
  const float* tp     = (const float*)d_in[3];
  const float* x_mean = (const float*)d_in[4];
  const float* h0     = (const float*)d_in[5];
  const float* w_gx   = (const float*)d_in[6];
  const float* h_gx   = (const float*)d_in[7];
  const float* w_gh   = (const float*)d_in[8];
  const float* h_gh   = (const float*)d_in[9];
  const float* w_t    = (const float*)d_in[10];
  const float* h_t    = (const float*)d_in[11];
  const float* W_ih   = (const float*)d_in[12];
  const float* W_hh   = (const float*)d_in[13];
  const float* b_ih   = (const float*)d_in[14];
  const float* b_hh   = (const float*)d_in[15];
  const float* W1     = (const float*)d_in[16];
  const float* b1     = (const float*)d_in[17];
  const float* W2     = (const float*)d_in[18];
  const float* b2     = (const float*)d_in[19];

  char* ws = (char*)d_ws;
  size_t off = 0;
  auto alloc = [&](size_t bytes) -> char* {
    char* p = ws + off;
    off = (off + bytes + 255) & ~(size_t)255;
    return p;
  };
  bf16* dt_bf = (bf16*)alloc((size_t)BT * I_DIM * 2);
  bf16* wgxT  = (bf16*)alloc((size_t)I_DIM * I_DIM * 2);
  bf16* wghT  = (bf16*)alloc((size_t)I_DIM * H_DIM * 2);
  bf16* wih   = (bf16*)alloc((size_t)G3H * I_DIM * 2);
  bf16* whh   = (bf16*)alloc((size_t)G3H * H_DIM * 2);
  float* gamma_h_ws = (float*)alloc((size_t)BT * H_DIM * 4);
  float* gi_ws      = (float*)alloc((size_t)BT * G3H * 4);

  float* out  = (float*)d_out;
  float* xhat = out + (size_t)T_DIM * B_DIM * 2;

  k_cvt<<<2048, 256, 0, stream>>>(dt, dt_bf, BT * I_DIM);
  k_cvt_t<<<64, 256, 0, stream>>>(w_gx, wgxT, I_DIM, I_DIM);
  k_cvt_t<<<128, 256, 0, stream>>>(w_gh, wghT, I_DIM, H_DIM);
  k_cvt<<<384, 256, 0, stream>>>(W_ih, wih, G3H * I_DIM);
  k_cvt<<<768, 256, 0, stream>>>(W_hh, whh, G3H * H_DIM);

  size_t shA = (size_t)8 * 16 * I_DIM * sizeof(bf16);  // 32 KB
  size_t shB = (size_t)(16 * H_DIM + 2 * 16 * H_DIM + 160) * sizeof(float) +
               (size_t)16 * H_DIM * sizeof(bf16);      // ~57 KB
  (void)hipFuncSetAttribute(reinterpret_cast<const void*>(k_pregate),
                            hipFuncAttributeMaxDynamicSharedMemorySize, (int)shA);
  (void)hipFuncSetAttribute(reinterpret_cast<const void*>(k_recur),
                            hipFuncAttributeMaxDynamicSharedMemorySize, (int)shB);

  k_pregate<<<BT / 128, 256, shA, stream>>>(x, mask, x_mean, h_gx, h_gh, b_ih,
                                            dt_bf, wgxT, wghT, wih,
                                            gamma_h_ws, gi_ws, xhat);

  k_recur<<<B_DIM / 16, 256, shB, stream>>>(tp, h0, w_t, h_t, b_hh, W1, b1, W2, b2,
                                            whh, gamma_h_ws, gi_ws, out);
}
